// GCN_grad_4836133175660
// MI455X (gfx1250) — compile-verified, matrix operand order checked by reading, not run on hardware
//
#include <hip/hip_runtime.h>

typedef __attribute__((ext_vector_type(16))) _Float16 v16h;
typedef __attribute__((ext_vector_type(8)))  _Float16 v8h;
typedef __attribute__((ext_vector_type(8)))  float    v8f;
typedef _Float16 h16;

#define XST 34  // xt row stride in floats (max width = 2 + 32)

// silu via hardware v_rcp_f32 (avoid IEEE div expansion)
__device__ __forceinline__ float siluf(float x) {
  return x * __builtin_amdgcn_rcpf(1.0f + __expf(-x));
}

__device__ __forceinline__ v8f wmma_f16(v16h a, v16h b, v8f c) {
  return __builtin_amdgcn_wmma_f32_16x16x32_f16(false, a, false, b, (short)0, c,
                                                false, false);
}

// ---------------------------------------------------------------------------
// A fragment from a zero-padded row-major LDS panel (ld = physical cols,
// multiple of 8; panel 16B aligned; k0 multiple of 32). Per the ISA 16-bit
// A layout, each lane's 16 halves are two contiguous 16B chunks:
//   ks [k0+hi*8, +8) and [k0+16+hi*8, +8)  ->  2x ds_load_b128, no branches.
// ---------------------------------------------------------------------------
__device__ __forceinline__ v16h load_fragA_pad(const h16* p, int ld, int k0) {
  const int lane = threadIdx.x & 31;
  const int row = lane & 15;
  const int hi = lane >> 4;
  const h16* base = p + row * ld + k0 + (hi ? 8 : 0);
  v8h lo = *(const v8h*)base;
  v8h hh = *(const v8h*)(base + 16);
  v16h f;
#pragma unroll
  for (int i = 0; i < 8; ++i) { f[i] = lo[i]; f[8 + i] = hh[i]; }
  return f;
}

// ---------------------------------------------------------------------------
// Packed-B weight staging: global f32 W[K x NCOLS] -> LDS in per-lane WMMA-B
// fragment order, zero-padded to KT*32 x NT*16. A fragment load is then one
// aligned 32-byte contiguous read per lane.
// layout: dst[(((nt*KT + kt)*32 + lane)*16 + j]
// ---------------------------------------------------------------------------
template <int K, int NCOLS, int KT, int NT, int BLK>
__device__ __forceinline__ void stage_packedB(h16* dstp, const float* W, int tid) {
  const int total = NT * KT * 512;
  for (int i = tid; i < total; i += BLK) {
    int j = i & 15;
    int lane = (i >> 4) & 31;
    int kt = (i >> 9) % KT;
    int nt = i / (KT * 512);
    int hi = lane >> 4;
    int col = nt * 16 + (lane & 15);
    int i2 = j >> 1;
    int k = kt * 32 + 2 * i2 + ((i2 >= 4) ? 8 : 0) + (hi ? 8 : 0) + (j & 1);
    float v = (k < K && col < NCOLS) ? W[k * NCOLS + col] : 0.f;
    dstp[i] = (h16)v;
  }
}

__device__ __forceinline__ v16h load_fragB_packed(const h16* p, int KT, int nt, int kt) {
  const int lane = threadIdx.x & 31;
  return *(const v16h*)(p + (((nt * KT + kt) * 32 + lane) << 4));
}

// Checked B loader (used only for Wc2: 128x1, raw storage, col 0 live)
__device__ __forceinline__ v16h load_fragB_vec(const h16* p, int k0, int kmax) {
  const int lane = threadIdx.x & 31;
  const int hi = lane >> 4;
  const bool cok = (lane & 15) == 0;
  v16h f;
#pragma unroll
  for (int i = 0; i < 8; ++i) {
    int k = k0 + 2 * i + ((i >= 4) ? 8 : 0) + (hi ? 8 : 0);
    f[2 * i]     = (cok && k     < kmax) ? p[k]     : (h16)0.f;
    f[2 * i + 1] = (cok && k + 1 < kmax) ? p[k + 1] : (h16)0.f;
  }
  return f;
}

// ---------------------------------------------------------------------------
// Fused edge kernel: one wave = 16 edges, all 4 edge GEMMs tile-resident.
//   m  = silu(silu(e_in @ We1 + be1) @ We2 + be2)   (E x 32)
//   cw = silu(m @ Wc1 + bc1) @ Wc2 + bc2            (E x 1)
//   atomics: coor[dst] += cw * rel ; m_i[dst] += m (if NEED_MI)
// Panel reuse: sA: e_in -> m ; sH: hidden -> c1.
// ---------------------------------------------------------------------------
template <int FIN, int EIN, int HID, int NW, bool NEED_MI>
__global__ void __launch_bounds__(NW * 32)
egnn_edge_kernel(const float* __restrict__ xt,
                 const int* __restrict__ src, const int* __restrict__ dst,
                 const float* __restrict__ eattr,
                 const float* We1, const float* be1,
                 const float* We2, const float* be2,
                 const float* Wc1, const float* bc1,
                 const float* Wc2, const float* bc2,
                 float* __restrict__ coor, float* __restrict__ m_i, int nE) {
  constexpr int KT1 = (EIN + 31) / 32;
  constexpr int KP1 = KT1 * 32;            // padded e_in width (>= 32 for m reuse)
  constexpr int NT1 = (HID + 15) / 16;
  constexpr int KT2 = (HID + 31) / 32;
  constexpr int KP2 = KT2 * 32;
  constexpr int HW = (KP2 > 128) ? KP2 : 128;  // hidden panel also hosts c1
  constexpr int BLK = NW * 32;

  __shared__ alignas(32) h16 sWe1p[NT1 * KT1 * 512];
  __shared__ alignas(32) h16 sWe2p[2 * KT2 * 512];
  __shared__ alignas(32) h16 sWc1p[8 * 512];
  __shared__ alignas(16) h16 sWc2[128];
  __shared__ float sbe1[HID];
  __shared__ float sbe2[32];
  __shared__ float sbc1[128];
  __shared__ float sbc2;
  __shared__ alignas(16) h16 sA[NW][16 * KP1];
  __shared__ alignas(16) h16 sH[NW][16 * HW];
  __shared__ float sRel[NW][32];
  __shared__ int   sDst[NW][16];

  const int tid = threadIdx.x;
  stage_packedB<EIN, HID, KT1, NT1, BLK>(sWe1p, We1, tid);
  stage_packedB<HID, 32, KT2, 2, BLK>(sWe2p, We2, tid);
  stage_packedB<32, 128, 1, 8, BLK>(sWc1p, Wc1, tid);
  for (int i = tid; i < 128; i += BLK) sWc2[i] = (h16)Wc2[i];
  for (int i = tid; i < HID; i += BLK) sbe1[i] = be1[i];
  for (int i = tid; i < 32; i += BLK) sbe2[i] = be2[i];
  for (int i = tid; i < 128; i += BLK) sbc1[i] = bc1[i];
  if (tid == 0) sbc2 = bc2[0];
  __syncthreads();

  const int w = tid >> 5;
  const int lane = tid & 31;
  const int tile = blockIdx.x * NW + w;
  if (tile * 16 >= nE) return;  // wave-uniform
  const int row = lane & 15;
  const int hi = lane >> 4;

  // ---- gather e_in = [feats[dst], feats[src], edge_attr, rel_dist] ---------
  {
    long e = (long)tile * 16 + row;
    bool valid = e < (long)nE;
    int s = valid ? src[e] : 0;
    int d = valid ? dst[e] : 0;
    if (!hi) {
      sDst[w][row] = valid ? d : -1;
      float cs0 = xt[(size_t)s * XST + 0], cs1 = xt[(size_t)s * XST + 1];
      float cd0 = xt[(size_t)d * XST + 0], cd1 = xt[(size_t)d * XST + 1];
      float r0 = valid ? (cs0 - cd0) : 0.f;
      float r1 = valid ? (cs1 - cd1) : 0.f;
      sRel[w][row * 2 + 0] = r0;
      sRel[w][row * 2 + 1] = r1;
#pragma unroll
      for (int j = 0; j < FIN; ++j)
        sA[w][row * KP1 + j] = valid ? (h16)xt[(size_t)d * XST + 2 + j] : (h16)0.f;
      sA[w][row * KP1 + 2 * FIN] = valid ? (h16)eattr[e] : (h16)0.f;
      sA[w][row * KP1 + 2 * FIN + 1] = (h16)(r0 * r0 + r1 * r1);
    } else {
#pragma unroll
      for (int j = 0; j < FIN; ++j)
        sA[w][row * KP1 + FIN + j] = valid ? (h16)xt[(size_t)s * XST + 2 + j] : (h16)0.f;
    }
    // zero-fill K padding so fragment loads are unconditional
    for (int t = lane; t < 16 * (KP1 - EIN); t += 32) {
      int r = t / (KP1 - EIN), c = EIN + t % (KP1 - EIN);
      sA[w][r * KP1 + c] = (h16)0.f;
    }
    for (int t = lane; t < 16 * (KP2 - HID); t += 32) {
      int r = t / (KP2 - HID), c = HID + t % (KP2 - HID);
      sH[w][r * HW + c] = (h16)0.f;
    }
  }
  // per-wave panels only: same-wave LDS ops are in-order (DScnt waits inserted)

  // ---- GEMM1: hidden = silu(e_in @ We1 + be1) -------------------------------
#pragma unroll
  for (int nt = 0; nt < NT1; ++nt) {
    v8f c = {};
#pragma unroll
    for (int kt = 0; kt < KT1; ++kt) {
      v16h a = load_fragA_pad(sA[w], KP1, kt * 32);
      v16h b = load_fragB_packed(sWe1p, KT1, nt, kt);
      c = wmma_f16(a, b, c);
    }
    int col = nt * 16 + (lane & 15);
    if (col < HID) {  // keep the zero padding intact
#pragma unroll
      for (int j = 0; j < 8; ++j)
        sH[w][(j + hi * 8) * HW + col] = (h16)siluf(c[j] + sbe1[col]);
    }
  }

  // ---- GEMM2: m = silu(hidden @ We2 + be2); m -> sA (e_in dead) -------------
#pragma unroll
  for (int nt = 0; nt < 2; ++nt) {
    v8f c = {};
#pragma unroll
    for (int kt = 0; kt < KT2; ++kt) {
      v16h a = load_fragA_pad(sH[w], HW, kt * 32);
      v16h b = load_fragB_packed(sWe2p, KT2, nt, kt);
      c = wmma_f16(a, b, c);
    }
    int col = nt * 16 + (lane & 15);
#pragma unroll
    for (int j = 0; j < 8; ++j) {
      int rr = j + hi * 8;
      float v = siluf(c[j] + sbe2[col]);
      sA[w][rr * KP1 + col] = (h16)v;
      if (NEED_MI) {
        int dn = sDst[w][rr];
        if (dn >= 0) atomicAdd(&m_i[(size_t)dn * 32 + col], v);
      }
    }
  }

  // ---- GEMM3: c1 = silu(m @ Wc1 + bc1), 32 -> 128; c1 -> sH ------------------
  {
    v16h a = load_fragA_pad(sA[w], KP1, 0);  // same m fragment for all 8 tiles
#pragma unroll
    for (int nt = 0; nt < 8; ++nt) {
      v8f c = {};
      v16h b = load_fragB_packed(sWc1p, 1, nt, 0);
      c = wmma_f16(a, b, c);
      int col = nt * 16 + (lane & 15);
#pragma unroll
      for (int j = 0; j < 8; ++j)
        sH[w][(j + hi * 8) * HW + col] = (h16)siluf(c[j] + sbc1[col]);
    }
  }

  // ---- GEMM4: cw = c1 @ Wc2 + bc2 (N padded to 16, col 0 live) --------------
  {
    v8f c = {};
#pragma unroll
    for (int kt = 0; kt < 4; ++kt) {
      v16h a = load_fragA_pad(sH[w], HW, kt * 32);
      v16h b = load_fragB_vec(sWc2, kt * 32, 128);
      c = wmma_f16(a, b, c);
    }
    if ((lane & 15) == 0) {
#pragma unroll
      for (int j = 0; j < 8; ++j) {
        int rr = j + hi * 8;
        int dn = sDst[w][rr];
        if (dn >= 0) {
          float cw = c[j] + sbc2;
          atomicAdd(&coor[(size_t)dn * 2 + 0], cw * sRel[w][rr * 2 + 0]);
          atomicAdd(&coor[(size_t)dn * 2 + 1], cw * sRel[w][rr * 2 + 1]);
        }
      }
    }
  }
}

// ---------------------------------------------------------------------------
// Node kernel: h = silu([feats, m_i] @ Wn1 + bn1) @ Wn2 + bn2 -> xt_next
// ---------------------------------------------------------------------------
template <int FIN, int NW>
__global__ void __launch_bounds__(NW * 32)
egnn_node_kernel(const float* __restrict__ xt, const float* __restrict__ m_i,
                 const float* __restrict__ coor,
                 const float* Wn1, const float* bn1,
                 const float* Wn2, const float* bn2,
                 float* __restrict__ xt_next, int nN) {
  constexpr int KN = FIN + 32;
  constexpr int KTN = (KN + 31) / 32;
  constexpr int KPN = KTN * 32;
  constexpr int BLK = NW * 32;

  __shared__ alignas(32) h16 sWn1p[4 * KTN * 512];
  __shared__ alignas(32) h16 sWn2p[2 * 2 * 512];
  __shared__ float sbn1[64];
  __shared__ float sbn2[32];
  __shared__ alignas(16) h16 sA[NW][16 * KPN];
  __shared__ alignas(16) h16 sH[NW][16 * 64];

  const int tid = threadIdx.x;
  stage_packedB<KN, 64, KTN, 4, BLK>(sWn1p, Wn1, tid);
  stage_packedB<64, 32, 2, 2, BLK>(sWn2p, Wn2, tid);
  for (int i = tid; i < 64; i += BLK) sbn1[i] = bn1[i];
  for (int i = tid; i < 32; i += BLK) sbn2[i] = bn2[i];
  __syncthreads();

  const int w = tid >> 5;
  const int lane = tid & 31;
  const int tile = blockIdx.x * NW + w;
  if (tile * 16 >= nN) return;
  const int row = lane & 15;
  const int hi = lane >> 4;

  {  // gather [feats, m_i]; lane halves split columns; zero-pad K tail
    int v = tile * 16 + row;
    bool valid = v < nN;
    constexpr int KH = KN / 2;
    int j0 = hi ? KH : 0;
    int j1 = hi ? KN : KH;
    for (int j = j0; j < j1; ++j) {
      float val = 0.f;
      if (valid)
        val = (j < FIN) ? xt[(size_t)v * XST + 2 + j]
                        : m_i[(size_t)v * 32 + (j - FIN)];
      sA[w][row * KPN + j] = (h16)val;
    }
    for (int t = lane; t < 16 * (KPN - KN); t += 32) {
      int r = t / (KPN - KN), c = KN + t % (KPN - KN);
      sA[w][r * KPN + c] = (h16)0.f;
    }
  }

#pragma unroll
  for (int nt = 0; nt < 4; ++nt) {
    v8f c = {};
#pragma unroll
    for (int kt = 0; kt < KTN; ++kt) {
      v16h a = load_fragA_pad(sA[w], KPN, kt * 32);
      v16h b = load_fragB_packed(sWn1p, KTN, nt, kt);
      c = wmma_f16(a, b, c);
    }
    int col = nt * 16 + (lane & 15);
#pragma unroll
    for (int j = 0; j < 8; ++j)
      sH[w][(j + hi * 8) * 64 + col] = (h16)siluf(c[j] + sbn1[col]);
  }

#pragma unroll
  for (int nt = 0; nt < 2; ++nt) {
    v8f c = {};
#pragma unroll
    for (int kt = 0; kt < 2; ++kt) {
      v16h a = load_fragA_pad(sH[w], 64, kt * 32);
      v16h b = load_fragB_packed(sWn2p, 2, nt, kt);
      c = wmma_f16(a, b, c);
    }
    int col = nt * 16 + (lane & 15);
#pragma unroll
    for (int j = 0; j < 8; ++j) {
      int rr = j + hi * 8;
      int v = tile * 16 + rr;
      if (v < nN) xt_next[(size_t)v * XST + 2 + col] = c[j] + sbn2[col];
    }
  }

  if ((lane & 15) == 0) {  // carry coords into xt_next
#pragma unroll
    for (int j = 0; j < 8; ++j) {
      int rr = j + hi * 8;
      int v = tile * 16 + rr;
      if (v < nN) {
        xt_next[(size_t)v * XST + 0] = coor[(size_t)v * 2 + 0];
        xt_next[(size_t)v * XST + 1] = coor[(size_t)v * 2 + 1];
      }
    }
  }
}

// ------------------------------ utility kernels ----------------------------
__global__ void build_xt0(const float* __restrict__ x, const float* __restrict__ pos,
                          float* __restrict__ xt, int nN) {
  int i = blockIdx.x * blockDim.x + threadIdx.x;
  if (i < nN) {
    xt[(size_t)i * XST + 0] = pos[(size_t)i * 2 + 0];
    xt[(size_t)i * XST + 1] = pos[(size_t)i * 2 + 1];
    xt[(size_t)i * XST + 2] = x[(size_t)i * 2 + 0];
    xt[(size_t)i * XST + 3] = x[(size_t)i * 2 + 1];
  }
}

__global__ void init_coor(const float* __restrict__ xt, float* __restrict__ coor, int nN) {
  int i = blockIdx.x * blockDim.x + threadIdx.x;
  if (i < nN * 2) coor[i] = xt[(size_t)(i >> 1) * XST + (i & 1)];
}

__global__ void zero_mi(float* __restrict__ m_i, int nN) {
  int i = blockIdx.x * blockDim.x + threadIdx.x;
  if (i < nN * 32) m_i[i] = 0.f;
}

__global__ void copy_out(const float* __restrict__ coor, float* __restrict__ out, int nN) {
  int i = blockIdx.x * blockDim.x + threadIdx.x;
  if (i < nN * 2) out[i] = coor[i];
}

// ---------------------------------------------------------------------------
extern "C" void kernel_launch(void* const* d_in, const int* in_sizes, int n_in,
                              void* d_out, int out_size, void* d_ws, size_t ws_size,
                              hipStream_t stream) {
  const float* x   = (const float*)d_in[0];
  const int*   ei  = (const int*)d_in[1];
  const float* ea  = (const float*)d_in[2];
  const float* pos = (const float*)d_in[4];
  const int N = in_sizes[0] / 2;   // x is (N, 2)
  const int E = in_sizes[2];       // edge_attr is (E, 1)
  const int* srcp = ei;            // edge_index[0]
  const int* dstp = ei + E;        // edge_index[1]

  const float* P0[12];
  const float* P1[12];
  for (int i = 0; i < 12; ++i) {
    P0[i] = (const float*)d_in[5 + i];
    P1[i] = (const float*)d_in[17 + i];
  }

  float* xtA  = (float*)d_ws;               // N x 34
  float* xtB  = xtA + (size_t)N * XST;      // N x 34
  float* coor = xtB + (size_t)N * XST;      // N x 2
  float* mi   = coor + (size_t)N * 2;       // N x 32

  const int eTiles = (E + 15) / 16;
  const int eBlocks = (eTiles + 1) / 2;     // 2 waves / block
  const int nTiles = (N + 15) / 16;
  const int nBlocks = (nTiles + 3) / 4;     // 4 waves / block

  build_xt0<<<(N + 255) / 256, 256, 0, stream>>>(x, pos, xtA, N);

  // ---- layer 0 ----
  init_coor<<<(N * 2 + 255) / 256, 256, 0, stream>>>(xtA, coor, N);
  zero_mi<<<(N * 32 + 255) / 256, 256, 0, stream>>>(mi, N);
  egnn_edge_kernel<2, 6, 12, 2, true><<<eBlocks, 64, 0, stream>>>(
      xtA, srcp, dstp, ea, P0[0], P0[1], P0[2], P0[3], P0[4], P0[5], P0[6], P0[7],
      coor, mi, E);
  egnn_node_kernel<2, 4><<<nBlocks, 128, 0, stream>>>(
      xtA, mi, coor, P0[8], P0[9], P0[10], P0[11], xtB, N);

  // ---- layer 1 (output needs coords only: node MLP + m_i segment-sum elided) ----
  init_coor<<<(N * 2 + 255) / 256, 256, 0, stream>>>(xtB, coor, N);
  egnn_edge_kernel<32, 66, 132, 2, false><<<eBlocks, 64, 0, stream>>>(
      xtB, srcp, dstp, ea, P1[0], P1[1], P1[2], P1[3], P1[4], P1[5], P1[6], P1[7],
      coor, mi, E);

  copy_out<<<(N * 2 + 255) / 256, 256, 0, stream>>>(coor, (float*)d_out, N);
}
